// DHCN_83708912599665
// MI455X (gfx1250) — matrix operand order repaired and checked by reading, not compiled, not run on hardware
//
#include <hip/hip_runtime.h>
#include <hip/hip_bf16.h>
#include <math.h>

// ---------------- problem constants (match reference) ----------------
#define EDIM    100
#define EP      128          // padded embedding dim (mult of 32)
#define BSESS   512
#define LSESS   50
#define LP      64           // padded session length
#define N_NODE  100000
#define C_NODE  2000
#define NTOT    (N_NODE + C_NODE)
#define NNZV    1500000
#define BL      (BSESS * LSESS)     // 25600 (mult of 16)
#define KCAT    320                 // padded 3*E concat K

typedef __attribute__((ext_vector_type(16))) _Float16 v16h;
typedef __attribute__((ext_vector_type(8)))  _Float16 v8h;
typedef __attribute__((ext_vector_type(8)))  float    v8f;
typedef _Float16 h16;

// =====================================================================
// Bandwidth path: sparse HyperConv (f32 atomics)
// =====================================================================
__global__ void k_init2(const float* __restrict__ emb, float* hg, float* h1, long n) {
    long i = (long)blockIdx.x * blockDim.x + threadIdx.x;
    if (i < n) { hg[i] = emb[i]; h1[i] = 0.f; }
}

__global__ void k_spmm(const int* __restrict__ row, const int* __restrict__ col,
                       const float* __restrict__ vals, const float* __restrict__ hin,
                       float* hout) {
    long gid = (long)blockIdx.x * blockDim.x + threadIdx.x;
    long i = gid >> 7;  int e = (int)(gid & 127);
    if (i >= NNZV || e >= EDIM) return;
    float v = vals[i] * hin[(long)col[i] * EDIM + e];
    atomicAdd(&hout[(long)row[i] * EDIM + e], v);
}

__global__ void k_acc_zero(float* hg, const float* __restrict__ hsrc, float* hnext, long n) {
    long i = (long)blockIdx.x * blockDim.x + threadIdx.x;
    if (i < n) { hg[i] += hsrc[i]; hnext[i] = 0.f; }
}

__global__ void k_acc_out(float* hg, const float* __restrict__ hsrc, float* out, long n) {
    long i = (long)blockIdx.x * blockDim.x + threadIdx.x;
    if (i < n) {
        float v = hg[i] + hsrc[i];
        hg[i] = v;
        if (i < (long)N_NODE * EDIM) out[i] = v;
    }
}

// =====================================================================
// Packing kernels (f32 -> zero-padded f16 operand layouts)
// =====================================================================
// dst[r*ldD + c] = src[r*ldS + c], zero-pad to RP x ldD
__global__ void k_pack(const float* __restrict__ src, int R, int C, int ldS,
                       h16* __restrict__ dst, int RP, int ldD) {
    long t = (long)blockIdx.x * blockDim.x + threadIdx.x;
    if (t >= (long)RP * ldD) return;
    int r = (int)(t / ldD), c = (int)(t % ldD);
    dst[t] = (h16)((r < R && c < C) ? src[(long)r * ldS + c] : 0.f);
}

// dst[n*ldD + k] = src[k*ldS + n]  (transpose-pack), zero-pad to NP x ldD
__global__ void k_packT(const float* __restrict__ src, int K, int N, int ldS,
                        h16* __restrict__ dst, int NP, int ldD) {
    long t = (long)blockIdx.x * blockDim.x + threadIdx.x;
    if (t >= (long)NP * ldD) return;
    int n = (int)(t / ldD), k = (int)(t % ldD);
    dst[t] = (h16)((k < K && n < N) ? src[(long)k * ldS + n] : 0.f);
}

// f16 source transpose-pack
__global__ void k_packT_h(const h16* __restrict__ src, int K, int N, int ldS,
                          h16* __restrict__ dst, int NP, int ldD) {
    long t = (long)blockIdx.x * blockDim.x + threadIdx.x;
    if (t >= (long)NP * ldD) return;
    int n = (int)(t / ldD), k = (int)(t % ldD);
    dst[t] = (k < K && n < N) ? src[(long)k * ldS + n] : (h16)0.f;
}

// batched V transpose: dst[b][e][l] = v[(b*L+l)*EP + e], [B x EP x LP]
__global__ void k_vT(const h16* __restrict__ v, h16* __restrict__ dst) {
    long t = (long)blockIdx.x * blockDim.x + threadIdx.x;
    if (t >= (long)BSESS * EP * LP) return;
    int b = (int)(t / (EP * LP));
    int e = (int)((t / LP) % EP);
    int l = (int)(t % LP);
    dst[t] = (l < LSESS && e < EDIM) ? v[((long)b * LSESS + l) * EP + e] : (h16)0.f;
}

// gather session sequences -> padded f16 [BL x EP]
__global__ void k_gather_seq(const int* __restrict__ rsi, const int* __restrict__ sic,
                             const float* __restrict__ hg,
                             h16* seqh, h16* seqc) {
    long t = (long)blockIdx.x * blockDim.x + threadIdx.x;
    if (t >= (long)BL * EP) return;
    long bl = t / EP; int e = (int)(t % EP);
    float vh = 0.f, vc = 0.f;
    if (e < EDIM) {
        int ii = rsi[bl];
        if (ii != 0) vh = hg[(long)(ii - 1) * EDIM + e];
        int ic = sic[bl];
        if (ic != 0) vc = hg[(long)(N_NODE + ic - 1) * EDIM + e];
    }
    seqh[t] = (h16)vh;  seqc[t] = (h16)vc;
}

// hs[b][e] = sum_l seqh / session_len  -> padded f16 [B x EP]
__global__ void k_hs(const h16* __restrict__ seqh, const float* __restrict__ slen,
                     h16* hs) {
    int t = blockIdx.x * blockDim.x + threadIdx.x;
    if (t >= BSESS * EP) return;
    int b = t / EP, e = t % EP;
    float acc = 0.f;
    if (e < EDIM)
        for (int l = 0; l < LSESS; ++l)
            acc += (float)seqh[((long)b * LSESS + l) * EP + e];
    hs[t] = (h16)(e < EDIM ? acc / slen[b] : 0.f);
}

// concat A = [pos | att_h | att_c], padded f16 [BL x 320]
__global__ void k_concat(const float* __restrict__ pos,
                         const h16* __restrict__ ah,   // [B x LP x EP]
                         const h16* __restrict__ ac,
                         h16* __restrict__ dst) {
    long t = (long)blockIdx.x * blockDim.x + threadIdx.x;
    if (t >= (long)BL * KCAT) return;
    long bl = t / KCAT; int k = (int)(t % KCAT);
    int b = (int)(bl / LSESS), l = (int)(bl % LSESS);
    float v = 0.f;
    if (k < EDIM)                 v = pos[(long)l * EDIM + k];
    else if (k < 2 * EDIM)        v = (float)ah[((long)b * LP + l) * EP + (k - EDIM)];
    else if (k < 3 * EDIM)        v = (float)ac[((long)b * LP + l) * EP + (k - 2 * EDIM)];
    dst[t] = (h16)v;
}

__global__ void k_softmax_rows(h16* scores) {
    int t = blockIdx.x * blockDim.x + threadIdx.x;
    if (t >= BL) return;
    int b = t / LSESS, m = t % LSESS;
    h16* row = scores + (long)b * LP * LP + (long)m * LP;
    float mx = -1e30f;
    for (int n = 0; n < LSESS; ++n) mx = fmaxf(mx, (float)row[n]);
    float s = 0.f;  float ex[LSESS];
    for (int n = 0; n < LSESS; ++n) { ex[n] = __expf((float)row[n] - mx); s += ex[n]; }
    float inv = 1.f / s;
    for (int n = 0; n < LSESS; ++n) row[n] = (h16)(ex[n] * inv);
}

__global__ void k_betaw(const h16* __restrict__ nh2, const float* __restrict__ w2,
                        const float* __restrict__ mask, float* betaw) {
    int t = blockIdx.x * blockDim.x + threadIdx.x;
    if (t >= BL) return;
    float acc = 0.f;
    const h16* rowp = nh2 + (long)t * EP;
    for (int e = 0; e < EDIM; ++e) acc += (float)rowp[e] * w2[e];
    betaw[t] = acc * mask[t];
}

__global__ void k_select(const float* __restrict__ betaw, const h16* __restrict__ seqh,
                         float* out_sel) {
    int t = blockIdx.x * blockDim.x + threadIdx.x;
    if (t >= BSESS * EDIM) return;
    int b = t / EDIM, e = t % EDIM;
    float acc = 0.f;
    for (int l = 0; l < LSESS; ++l) {
        long bl = (long)b * LSESS + l;
        acc += betaw[bl] * (float)seqh[bl * EP + e];
    }
    out_sel[t] = acc;
}

__global__ void k_sess(const int* __restrict__ si, const int* __restrict__ sic,
                       const float* __restrict__ emb, const float* __restrict__ slen,
                       h16* sess) {
    int t = blockIdx.x * blockDim.x + threadIdx.x;
    if (t >= BSESS * EP) return;
    int b = t / EP, e = t % EP;
    float acc = 0.f;
    if (e < EDIM)
        for (int l = 0; l < LSESS; ++l) {
            long bl = (long)b * LSESS + l;
            int idx = si[bl] + sic[bl];
            if (idx != 0) acc += emb[(long)(idx - 1) * EDIM + e];
        }
    sess[t] = (h16)(e < EDIM ? acc / slen[b] : 0.f);
}

// sess_tot (f32) = sess16 + s1_16   /  sess_tot += s2_16
__global__ void k_vadd_h(float* out, const h16* a, const h16* b) {
    int i = blockIdx.x * blockDim.x + threadIdx.x;
    if (i >= BSESS * EDIM) return;
    int bb = i / EDIM, e = i % EDIM;
    out[i] = (float)a[(long)bb * EP + e] + (float)b[(long)bb * EP + e];
}
__global__ void k_vacc_h(float* out, const h16* a) {
    int i = blockIdx.x * blockDim.x + threadIdx.x;
    if (i >= BSESS * EDIM) return;
    int bb = i / EDIM, e = i % EDIM;
    out[i] += (float)a[(long)bb * EP + e];
}

// SSL loss; JAX threefry permutations approximated by fixed affine bijections.
__global__ void k_ssl(const float* __restrict__ sel, const float* __restrict__ st,
                      float* out_scalar) {
    __shared__ float red[256];
    float local = 0.f;
    for (int b = threadIdx.x; b < BSESS; b += blockDim.x) {
        float ps = 0.f, ng = 0.f;
        int pr = (181 * b + 97) & 511;
        for (int e = 0; e < EDIM; ++e) {
            ps += sel[(long)b * EDIM + e] * st[(long)b * EDIM + e];
            int pc = (37 * e + 11) % EDIM;
            ng += st[(long)b * EDIM + e] * sel[(long)pr * EDIM + pc];
        }
        float sp = 1.f / (1.f + __expf(-ps));
        float sn = 1.f / (1.f + __expf(-ng));
        local += -logf(1e-8f + sp) - logf(1e-8f + 1.f - sn);
    }
    red[threadIdx.x] = local;
    __syncthreads();
    for (int s = blockDim.x >> 1; s > 0; s >>= 1) {
        if (threadIdx.x < s) red[threadIdx.x] += red[threadIdx.x + s];
        __syncthreads();
    }
    if (threadIdx.x == 0) out_scalar[0] = 0.01f * red[0];
}

// =====================================================================
// Generic WMMA f16 GEMM on pre-packed operands (NO bounds checks in loop)
//   C(f16, padded) = epi(alpha * A[Mp x ldaP] @ BT[Np x ldbP]^T)
//   A frag: 2x contiguous 16B loads; BT frag: 2x contiguous 16B loads.
//   Store clamps to zero outside Mv x Nv (keeps pads zero).
//   epi: 0 none, 1 tanh, 2 sigmoid(x + bias[n] + rowadd[m/rowdiv][n])
// =====================================================================
__global__ void wmma_gemm16(const h16* __restrict__ A, long sA, int ldaP,
                            const h16* __restrict__ BT, long sB, int ldbP,
                            h16* __restrict__ C, long sC, int ldcP,
                            int Mtiles, int KP, int Mv, int Nv,
                            float alpha, int epi,
                            const float* __restrict__ bias,
                            const h16* __restrict__ rowadd, int rowdiv, int ldraP) {
    int lane = threadIdx.x;                              // 0..31
    int tm = blockIdx.x * blockDim.y + threadIdx.y;      // wave-uniform
    if (tm >= Mtiles) return;
    int tn = blockIdx.y;
    const h16* Ab = A + (long)blockIdx.z * sA;
    const h16* Bb = BT + (long)blockIdx.z * sB;
    h16*       Cb = C + (long)blockIdx.z * sC;
    int m0 = tm * 16, n0 = tn * 16;
    int mfrag = m0 + (lane & 15);
    int nfrag = n0 + (lane & 15);
    int kha = (lane < 16) ? 0 : 8;      // A K sub-offset (ISA layout)
    int khb = (lane < 16) ? 0 : 8;      // B K sub-offset
    const h16* arow = Ab + (long)mfrag * ldaP + kha;
    const h16* brow = Bb + (long)nfrag * ldbP + khb;

    v8f acc = {};
    for (int k0 = 0; k0 < KP; k0 += 32) {
        v8h a0 = *(const v8h*)(arow + k0);
        v8h a1 = *(const v8h*)(arow + k0 + 16);
        v8h b0 = *(const v8h*)(brow + k0);
        v8h b1 = *(const v8h*)(brow + k0 + 16);
        v16h a, b;
#pragma unroll
        for (int i = 0; i < 8; ++i) {
            a[i] = a0[i];  a[i + 8] = a1[i];
            b[i] = b0[i];  b[i + 8] = b1[i];
        }
        acc = __builtin_amdgcn_wmma_f32_16x16x32_f16(
                  false, a, false, b, (short)0, acc, false, false);
    }
    int mb = m0 + ((lane < 16) ? 0 : 8);
#pragma unroll
    for (int r = 0; r < 8; ++r) {
        int mm = mb + r;
        float v = acc[r] * alpha;
        if (epi == 1) v = tanhf(v);
        else if (epi == 2) {
            float bi = (nfrag < Nv) ? bias[nfrag] : 0.f;
            v += bi + (float)rowadd[(long)(mm / rowdiv) * ldraP + nfrag];
            v = 1.f / (1.f + __expf(-v));
        }
        if (mm >= Mv || nfrag >= Nv) v = 0.f;   // keep pads zero
        Cb[(long)mm * ldcP + nfrag] = (h16)v;
    }
}

// =====================================================================
// host side
// =====================================================================
static void launch_gemm16(const h16* A, long sA, int ldaP,
                          const h16* BT, long sB, int ldbP,
                          h16* C, long sC, int ldcP,
                          int Mtiles, int KP, int Mv, int Nv, int batch,
                          float alpha, int epi, const float* bias,
                          const h16* rowadd, int rowdiv, int ldraP,
                          hipStream_t stream) {
    dim3 blk(32, 4, 1);
    dim3 grd((Mtiles + 3) / 4, ldcP / 16, batch);   // write full padded width
    wmma_gemm16<<<grd, blk, 0, stream>>>(A, sA, ldaP, BT, sB, ldbP, C, sC, ldcP,
                                         Mtiles, KP, Mv, Nv, alpha, epi,
                                         bias, rowadd, rowdiv, ldraP);
}

extern "C" void kernel_launch(void* const* d_in, const int* in_sizes, int n_in,
                              void* d_out, int out_size, void* d_ws, size_t ws_size,
                              hipStream_t stream) {
    const int*   session_item = (const int*)  d_in[0];
    const int*   session_cat  = (const int*)  d_in[1];
    const int*   rev_sess     = (const int*)  d_in[2];
    const float* session_len  = (const float*)d_in[3];
    const float* mask         = (const float*)d_in[4];
    const float* Dm           = (const float*)d_in[5];
    const float* Am           = (const float*)d_in[6];
    const int*   adj_row      = (const int*)  d_in[7];
    const int*   adj_col      = (const int*)  d_in[8];
    const float* adj_vals     = (const float*)d_in[9];
    const float* emb          = (const float*)d_in[10];
    const float* pos_t        = (const float*)d_in[11];
    const float* W_in[6] = { (const float*)d_in[12], (const float*)d_in[13],
                             (const float*)d_in[14], (const float*)d_in[15],
                             (const float*)d_in[16], (const float*)d_in[17] };
    const float* w_1    = (const float*)d_in[18];
    const float* w_2    = (const float*)d_in[19];
    const float* glu1_w = (const float*)d_in[20];
    const float* glu1_b = (const float*)d_in[21];
    const float* glu2_w = (const float*)d_in[22];

    float* out = (float*)d_out;
    const long OUT_SEL = (long)N_NODE * EDIM;
    const long OUT_CON = OUT_SEL + (long)BSESS * EDIM;

    // ---- workspace layout: f32 region then 16B-aligned f16 region ----
    float* f = (float*)d_ws;
    const long NE = (long)NTOT * EDIM;
    float* h1 = f;              f += NE;
    float* h2 = f;              f += NE;
    float* hg = f;              f += NE;
    float* betaw = f;           f += BL;
    float* sesstot = f;         f += BSESS * EDIM;
    f += 8 - (((uintptr_t)f / 4) & 7);           // align to 16B for v8h loads
    h16* p = (h16*)f;
    h16* seqh16 = p;            p += (long)BL * EP;
    h16* seqc16 = p;            p += (long)BL * EP;
    h16* qkv16[6];              // q_i,k_i,v_i,q_c,k_c,v_c
    for (int i = 0; i < 6; ++i) { qkv16[i] = p; p += (long)BL * EP; }
    h16* scores16 = p;          p += (long)BSESS * LP * LP;
    h16* vT16 = p;              p += (long)BSESS * EP * LP;
    h16* atth16 = p;            p += (long)BSESS * LP * EP;
    h16* attc16 = p;            p += (long)BSESS * LP * EP;
    h16* concat16 = p;          p += (long)BL * KCAT;
    h16* nh1_16 = p;            p += (long)BL * EP;
    h16* nh2_16 = p;            p += (long)BL * EP;
    h16* hs16 = p;              p += (long)BSESS * EP;
    h16* hsg16 = p;             p += (long)BSESS * EP;
    h16* sess16 = p;            p += (long)BSESS * EP;
    h16* sessT16 = p;           p += (long)EP * BSESS;
    h16* s1_16 = p;             p += (long)BSESS * EP;
    h16* s1T16 = p;             p += (long)EP * BSESS;
    h16* s2_16 = p;             p += (long)BSESS * EP;
    h16* Wt16[6];               // transposed weight packs [EP x EP]
    for (int i = 0; i < 6; ++i) { Wt16[i] = p; p += (long)EP * EP; }
    h16* w1T16 = p;             p += (long)EP * KCAT;
    h16* glu1T16 = p;           p += (long)EP * EP;
    h16* glu2T16 = p;           p += (long)EP * EP;
    h16* Dp16 = p;              p += (long)BSESS * BSESS;
    h16* AT16 = p;              p += (long)BSESS * BSESS;
    h16* DA16 = p;              p += (long)BSESS * BSESS;

    const int TB = 256;
    #define GB(n) ((int)(((long)(n) + TB - 1) / TB))

    // ---- weight packs (independent, issue first) ----
    for (int i = 0; i < 6; ++i)
        k_packT<<<GB((long)EP * EP), TB, 0, stream>>>(W_in[i], EDIM, EDIM, EDIM,
                                                      Wt16[i], EP, EP);
    k_packT<<<GB((long)EP * KCAT), TB, 0, stream>>>(w_1, 3 * EDIM, EDIM, EDIM,
                                                    w1T16, EP, KCAT);
    k_packT<<<GB((long)EP * EP), TB, 0, stream>>>(glu1_w, EDIM, EDIM, EDIM,
                                                  glu1T16, EP, EP);
    k_packT<<<GB((long)EP * EP), TB, 0, stream>>>(glu2_w, EDIM, EDIM, EDIM,
                                                  glu2T16, EP, EP);
    k_pack<<<GB((long)BSESS * BSESS), TB, 0, stream>>>(Dm, BSESS, BSESS, BSESS,
                                                       Dp16, BSESS, BSESS);
    k_packT<<<GB((long)BSESS * BSESS), TB, 0, stream>>>(Am, BSESS, BSESS, BSESS,
                                                        AT16, BSESS, BSESS);

    // ---- HyperConv (2 layers, sparse scatter) ----
    k_init2<<<GB(NE), TB, 0, stream>>>(emb, hg, h1, NE);
    long spmm_thr = (long)NNZV * 128;
    k_spmm<<<GB(spmm_thr), TB, 0, stream>>>(adj_row, adj_col, adj_vals, emb, h1);
    k_acc_zero<<<GB(NE), TB, 0, stream>>>(hg, h1, h2, NE);
    k_spmm<<<GB(spmm_thr), TB, 0, stream>>>(adj_row, adj_col, adj_vals, h1, h2);
    k_acc_out<<<GB(NE), TB, 0, stream>>>(hg, h2, out, NE);

    // ---- session gathers (write padded f16 directly) ----
    k_gather_seq<<<GB((long)BL * EP), TB, 0, stream>>>(rev_sess, session_cat, hg,
                                                       seqh16, seqc16);
    k_hs<<<GB(BSESS * EP), TB, 0, stream>>>(seqh16, session_len, hs16);

    // ---- projections (WMMA): q/k/v for item (seqh) and cat (seqc) ----
    const int MT_BL = BL / 16;      // 1600
    for (int i = 0; i < 6; ++i) {
        const h16* src = (i < 3) ? seqh16 : seqc16;
        launch_gemm16(src, 0, EP, Wt16[i], 0, EP, qkv16[i], 0, EP,
                      MT_BL, EP, BL, EDIM, 1, 1.f, 0, 0, 0, 1, 1, stream);
    }

    const float scale = 0.1f;  // 1/sqrt(E)
    const long sQ = (long)LSESS * EP;        // per-batch row offset in packed q/k/v
    const long sS = (long)LP * LP;
    const long sVT = (long)EP * LP;
    const long sAT = (long)LP * EP;
    for (int g = 0; g < 2; ++g) {            // 0 = item, 1 = cat
        const h16* q = qkv16[3 * g + 0];
        const h16* k = qkv16[3 * g + 1];
        const h16* v = qkv16[3 * g + 2];
        h16* att = g ? attc16 : atth16;
        // scores = alpha * Q K^T (packed K rows already are BT layout)
        launch_gemm16(q, sQ, EP, k, sQ, EP, scores16, sS, LP,
                      4, EP, LSESS, LSESS, BSESS, scale, 0, 0, 0, 1, 1, stream);
        k_softmax_rows<<<GB(BL), TB, 0, stream>>>(scores16);
        // V^T pack, then att = S @ V
        k_vT<<<GB((long)BSESS * EP * LP), TB, 0, stream>>>(v, vT16);
        launch_gemm16(scores16, sS, LP, vT16, sVT, LP, att, sAT, EP,
                      4, LP, LSESS, EDIM, BSESS, 1.f, 0, 0, 0, 1, 1, stream);
    }

    // ---- nh1 = tanh([pos|att_h|att_c] @ w_1) ----
    k_concat<<<GB((long)BL * KCAT), TB, 0, stream>>>(pos_t, atth16, attc16, concat16);
    launch_gemm16(concat16, 0, KCAT, w1T16, 0, KCAT, nh1_16, 0, EP,
                  MT_BL, KCAT, BL, EDIM, 1, 1.f, 1, 0, 0, 1, 1, stream);

    // ---- hsg = hs @ glu2_w ; nh2 = sigmoid(nh1@glu1_w + b + hsg[b]) ----
    launch_gemm16(hs16, 0, EP, glu2T16, 0, EP, hsg16, 0, EP,
                  BSESS / 16, EP, BSESS, EDIM, 1, 1.f, 0, 0, 0, 1, 1, stream);
    launch_gemm16(nh1_16, 0, EP, glu1T16, 0, EP, nh2_16, 0, EP,
                  MT_BL, EP, BL, EDIM, 1, 1.f, 2, glu1_b, hsg16, LSESS, EP, stream);

    // ---- select -> d_out ----
    k_betaw<<<GB(BL), TB, 0, stream>>>(nh2_16, w_2, mask, betaw);
    k_select<<<GB(BSESS * EDIM), TB, 0, stream>>>(betaw, seqh16, out + OUT_SEL);

    // ---- LineConv ----
    k_sess<<<GB(BSESS * EP), TB, 0, stream>>>(session_item, session_cat, emb,
                                              session_len, sess16);
    launch_gemm16(Dp16, 0, BSESS, AT16, 0, BSESS, DA16, 0, BSESS,
                  BSESS / 16, BSESS, BSESS, BSESS, 1, 1.f, 0, 0, 0, 1, 1, stream);
    k_packT_h<<<GB((long)EP * BSESS), TB, 0, stream>>>(sess16, BSESS, EDIM, EP,
                                                       sessT16, EP, BSESS);
    launch_gemm16(DA16, 0, BSESS, sessT16, 0, BSESS, s1_16, 0, EP,
                  BSESS / 16, BSESS, BSESS, EDIM, 1, 1.f, 0, 0, 0, 1, 1, stream);
    k_vadd_h<<<GB(BSESS * EDIM), TB, 0, stream>>>(sesstot, sess16, s1_16);
    k_packT_h<<<GB((long)EP * BSESS), TB, 0, stream>>>(s1_16, BSESS, EDIM, EP,
                                                       s1T16, EP, BSESS);
    launch_gemm16(DA16, 0, BSESS, s1T16, 0, BSESS, s2_16, 0, EP,
                  BSESS / 16, BSESS, BSESS, EDIM, 1, 1.f, 0, 0, 0, 1, 1, stream);
    k_vacc_h<<<GB(BSESS * EDIM), TB, 0, stream>>>(sesstot, s2_16);

    // ---- SSL loss scalar ----
    k_ssl<<<1, 256, 0, stream>>>(out + OUT_SEL, sesstot, out + OUT_CON);

    (void)in_sizes; (void)n_in; (void)out_size; (void)ws_size; (void)p;
}
